// VariousRNN_55980603736274
// MI455X (gfx1250) — matrix operand order repaired and checked by reading, not compiled, not run on hardware
//
#include <hip/hip_runtime.h>

// Problem dims (match reference)
#define TSEQ 128
#define BB   256
#define DD   512
#define HH   512
#define G4   2048   // 4*H

// LDS A-panel: 32 rows x 512 halves, rows padded by 8 halves (16B) to spread banks
#define APAD 8
#define AK   (DD + APAD)   // 520 halves per row

typedef __attribute__((ext_vector_type(16))) _Float16 v16h;
typedef __attribute__((ext_vector_type(8)))  _Float16 v8h;
typedef __attribute__((ext_vector_type(8)))  float    v8f;

// Types for the async global->LDS builtin (param types from hipcc diagnostic:
// 'int __attribute__((vector_size(16))) __device__ *' i.e. v4i in AS(1))
typedef int async_v4i __attribute__((vector_size(16)));
typedef __attribute__((address_space(1))) async_v4i ga_v4i;   // global AS
typedef __attribute__((address_space(3))) async_v4i la_v4i;   // LDS AS

// ---------------------------------------------------------------------------
// Helpers
// ---------------------------------------------------------------------------
static __device__ __forceinline__ float fast_sigmoid(float x) {
    return 1.0f / (1.0f + __expf(-x));
}
static __device__ __forceinline__ float fast_tanh(float x) {
    return 2.0f / (1.0f + __expf(-2.0f * x)) - 1.0f;
}

// Build the 16-bit A-operand (16x32 fragment) in contiguous k order (B loads match):
// lanes 0-15 : K = {k0..k0+15}
// lanes 16-31: K = {k0+16..k0+31}
static __device__ __forceinline__ v16h load_a_frag(const _Float16* rowp,
                                                   int k0, int l16) {
    v8h lo = *(const v8h*)(rowp + k0 + (l16 << 4));
    v8h hi = *(const v8h*)(rowp + k0 + 8 + (l16 << 4));
    return __builtin_shufflevector(lo, hi,
        0, 1, 2, 3, 4, 5, 6, 7, 8, 9, 10, 11, 12, 13, 14, 15);
}

// Stage a 32-row x 512-half A panel from global into LDS.
// Uses the CDNA5 async global->LDS path (ASYNCcnt) when available.
static __device__ __forceinline__ void stage_a_panel(const _Float16* __restrict__ gsrc,
                                                     _Float16* smem /* [32][AK] */) {
    const int tid = threadIdx.x;   // 256 threads, 2048 16B-chunks total -> 8 each
#if __has_builtin(__builtin_amdgcn_global_load_async_to_lds_b128)
#pragma unroll
    for (int i = 0; i < 8; ++i) {
        const int c   = tid + i * 256;
        const int row = c >> 6;          // 64 chunks per 512-half row
        const int ko  = (c & 63) << 3;   // half offset within row
        __builtin_amdgcn_global_load_async_to_lds_b128(
            (ga_v4i*)(gsrc + (size_t)row * DD + ko),
            (la_v4i*)(smem + row * AK + ko), 0, 0);
    }
#if __has_builtin(__builtin_amdgcn_s_wait_asynccnt)
    __builtin_amdgcn_s_wait_asynccnt(0);
#else
    asm volatile("s_wait_asynccnt 0" ::: "memory");
#endif
#else
#pragma unroll
    for (int i = 0; i < 8; ++i) {
        const int c   = tid + i * 256;
        const int row = c >> 6;
        const int ko  = (c & 63) << 3;
        *(v8h*)(smem + row * AK + ko) = *(const v8h*)(gsrc + (size_t)row * DD + ko);
    }
#endif
    __syncthreads();
}

// ---------------------------------------------------------------------------
// Prep: fp32 -> f16 conversion (for x)
// ---------------------------------------------------------------------------
__global__ void k_f32_to_f16(const float* __restrict__ src,
                             _Float16* __restrict__ dst, int n) {
    int i = blockIdx.x * blockDim.x + threadIdx.x;
    if (i < n) dst[i] = (_Float16)src[i];
}

// Prep: transpose [512, 2048] fp32 -> [2048, 512] f16 via LDS tile
__global__ void k_transpose_f16(const float* __restrict__ src,
                                _Float16* __restrict__ dst) {
    __shared__ float tile[16][17];
    int c = blockIdx.x * 16 + threadIdx.x;   // 0..2047
    int r = blockIdx.y * 16 + threadIdx.y;   // 0..511
    tile[threadIdx.y][threadIdx.x] = src[r * G4 + c];
    __syncthreads();
    int dr = blockIdx.x * 16 + threadIdx.y;  // n  (0..2047)
    int dc = blockIdx.y * 16 + threadIdx.x;  // k  (0..511)
    dst[(size_t)dr * DD + dc] = (_Float16)tile[threadIdx.x][threadIdx.y];
}

// Prep: zero initial h (f16, both dirs contiguous) and c (f32, both dirs)
__global__ void k_zero_state(_Float16* __restrict__ h0,
                             float* __restrict__ c0, int n) {
    int i = blockIdx.x * blockDim.x + threadIdx.x;
    if (i < n) { h0[i] = (_Float16)0.0f; c0[i] = 0.0f; }
}

// ---------------------------------------------------------------------------
// Input projection: xw = x @ W + b   (M = B*T = 32768, N = 4H = 2048, K = 512)
// grid = (N/512, M/32, 2 dirs), block = 256 (8 waves).
// Block stages a 32xK x-panel in LDS (async copy); each wave computes a
// 32x64 strip: 8 accumulators, 8 v_wmma per k-step.
// ---------------------------------------------------------------------------
__global__ void k_inproj(const _Float16* __restrict__ x16,
                         const _Float16* __restrict__ wt_f,
                         const _Float16* __restrict__ wt_b,
                         const float* __restrict__ b_f,
                         const float* __restrict__ b_b,
                         _Float16* __restrict__ xw_f,
                         _Float16* __restrict__ xw_b) {
    __shared__ _Float16 sa[32 * AK];

    const int dir = blockIdx.z;
    const _Float16* wt   = dir ? wt_b : wt_f;   // [2048, 512] f16 (pre-transposed)
    const float*    bias = dir ? b_b  : b_f;
    _Float16*       xw   = dir ? xw_b : xw_f;   // [32768, 2048] f16

    const int lane = threadIdx.x & 31;
    const int wave = threadIdx.x >> 5;
    const int l16  = lane >> 4;
    const int lm   = lane & 15;
    const int m0   = blockIdx.y * 32;
    const int n0   = blockIdx.x * 512 + wave * 64;

    stage_a_panel(x16 + (size_t)m0 * DD, sa);

    v8f acc[2][4] = {};
    const _Float16* ar0 = sa + lm * AK;          // M-tile 0 rows
    const _Float16* ar1 = sa + (16 + lm) * AK;   // M-tile 1 rows

#pragma unroll 4
    for (int k0 = 0; k0 < DD; k0 += 32) {
        v16h a0 = load_a_frag(ar0, k0, l16);
        v16h a1 = load_a_frag(ar1, k0, l16);
#pragma unroll
        for (int j = 0; j < 4; ++j) {
            // B (32x16): lane holds column n0+j*16+lm; 16 contiguous halves
            // from the [N,K] weight layout.
            const int colg = n0 + j * 16 + lm;
            v16h b = *(const v16h*)(wt + (size_t)colg * DD + k0 + (l16 << 4));
            acc[0][j] = __builtin_amdgcn_wmma_f32_16x16x32_f16(
                false, a0, false, b, (short)0, acc[0][j], false, false);
            acc[1][j] = __builtin_amdgcn_wmma_f32_16x16x32_f16(
                false, a1, false, b, (short)0, acc[1][j], false, false);
        }
    }

#pragma unroll
    for (int mt = 0; mt < 2; ++mt) {
        const int rbase = m0 + mt * 16 + (l16 << 3);
#pragma unroll
        for (int j = 0; j < 4; ++j) {
            const int colg = n0 + j * 16 + lm;
            const float bv = bias[colg];
#pragma unroll
            for (int r = 0; r < 8; ++r) {
                xw[(size_t)(rbase + r) * G4 + colg] = (_Float16)(acc[mt][j][r] + bv);
            }
        }
    }
}

// ---------------------------------------------------------------------------
// One recurrent step for both directions.
// grid = (HH/128, BB/32, 2 dirs), block = 256 (8 waves).
// Block stages a 32xH h-panel in LDS; each wave owns a 32(batch) x 16(hidden)
// tile with 4 gate accumulators per M-tile, fusing z = xw_t + h@U, the gate
// nonlinearities, the cell/hidden update and all stores.
// ---------------------------------------------------------------------------
__global__ void k_step(const _Float16* __restrict__ ut_f,
                       const _Float16* __restrict__ ut_b,
                       const _Float16* __restrict__ xw_f,
                       const _Float16* __restrict__ xw_b,
                       const _Float16* __restrict__ hin_f,
                       const _Float16* __restrict__ hin_b,
                       _Float16* __restrict__ hout_f,
                       _Float16* __restrict__ hout_b,
                       float* __restrict__ c_f,
                       float* __restrict__ c_b,
                       float* __restrict__ out, int s) {
    __shared__ _Float16 sa[32 * AK];

    const int dir = blockIdx.z;
    const _Float16* ut  = dir ? ut_b  : ut_f;   // [2048, 512] f16
    const _Float16* xw  = dir ? xw_b  : xw_f;
    const _Float16* hin = dir ? hin_b : hin_f;  // [256, 512] f16
    _Float16* hout = dir ? hout_b : hout_f;
    float*    cbuf = dir ? c_b    : c_f;
    const int t = dir ? (TSEQ - 1 - s) : s;

    const int lane = threadIdx.x & 31;
    const int wave = threadIdx.x >> 5;
    const int l16  = lane >> 4;
    const int lm   = lane & 15;
    const int m0   = blockIdx.y * 32;                 // batch tile
    const int nh   = blockIdx.x * 128 + wave * 16;    // hidden-col tile

    stage_a_panel(hin + (size_t)m0 * HH, sa);

    v8f acc[2][4] = {};
    const _Float16* ar0 = sa + lm * AK;
    const _Float16* ar1 = sa + (16 + lm) * AK;

#pragma unroll 4
    for (int k0 = 0; k0 < HH; k0 += 32) {
        v16h a0 = load_a_frag(ar0, k0, l16);
        v16h a1 = load_a_frag(ar1, k0, l16);
#pragma unroll
        for (int g = 0; g < 4; ++g) {
            const int browg = g * HH + nh + lm;       // row of U^T = output col
            v16h b = *(const v16h*)(ut + (size_t)browg * HH + k0 + (l16 << 4));
            acc[0][g] = __builtin_amdgcn_wmma_f32_16x16x32_f16(
                false, a0, false, b, (short)0, acc[0][g], false, false);
            acc[1][g] = __builtin_amdgcn_wmma_f32_16x16x32_f16(
                false, a1, false, b, (short)0, acc[1][g], false, false);
        }
    }

    const int col  = nh + lm;
    const bool last = (t == (dir ? 0 : TSEQ - 1));
#pragma unroll
    for (int mt = 0; mt < 2; ++mt) {
        const int rbase = m0 + mt * 16 + (l16 << 3);
#pragma unroll
        for (int r = 0; r < 8; ++r) {
            const int row = rbase + r;
            const size_t xrow = (size_t)(row * TSEQ + t) * G4;
            const float zi = acc[mt][0][r] + (float)xw[xrow + 0 * HH + col];
            const float zf = acc[mt][1][r] + (float)xw[xrow + 1 * HH + col];
            const float zg = acc[mt][2][r] + (float)xw[xrow + 2 * HH + col];
            const float zo = acc[mt][3][r] + (float)xw[xrow + 3 * HH + col];
            const float ig = fast_sigmoid(zi);
            const float fg = fast_sigmoid(zf);
            const float gg = fast_tanh(zg);
            const float og = fast_sigmoid(zo);
            const int cidx = row * HH + col;
            const float cv = fg * cbuf[cidx] + ig * gg;
            cbuf[cidx] = cv;
            const float hv = og * fast_tanh(cv);
            hout[cidx] = (_Float16)hv;
            out[(size_t)(row * TSEQ + t) * (2 * HH) + dir * HH + col] = hv;
            if (last)
                out[(size_t)(BB * TSEQ * 2 * HH) + (size_t)row * (2 * HH) + dir * HH + col] = hv;
        }
    }
}

// ---------------------------------------------------------------------------
// Host launcher
// ---------------------------------------------------------------------------
extern "C" void kernel_launch(void* const* d_in, const int* in_sizes, int n_in,
                              void* d_out, int out_size, void* d_ws, size_t ws_size,
                              hipStream_t stream) {
    (void)in_sizes; (void)n_in; (void)out_size; (void)ws_size;

    const float* x   = (const float*)d_in[0];   // [256,128,512]
    const float* W_f = (const float*)d_in[1];   // [512,2048]
    const float* U_f = (const float*)d_in[2];   // [512,2048]
    const float* b_f = (const float*)d_in[3];   // [2048]
    const float* W_b = (const float*)d_in[4];
    const float* U_b = (const float*)d_in[5];
    const float* b_b = (const float*)d_in[6];
    float* out = (float*)d_out;

    // Workspace layout (bytes)
    char* ws = (char*)d_ws;
    _Float16* x16  = (_Float16*)(ws + 0);              //  32 MB: x f16 [32768,512]
    _Float16* wt_f = (_Float16*)(ws + 33554432);       //   2 MB: W_f^T f16 [2048,512]
    _Float16* wt_b = (_Float16*)(ws + 35651584);
    _Float16* ut_f = (_Float16*)(ws + 37748736);
    _Float16* ut_b = (_Float16*)(ws + 39845888);
    _Float16* xw_f = (_Float16*)(ws + 41943040);       // 128 MB: xw f16 [32768,2048]
    _Float16* xw_b = (_Float16*)(ws + 176160768);
    _Float16* h0_f = (_Float16*)(ws + 310378496);      // 256 KB each
    _Float16* h0_b = (_Float16*)(ws + 310640640);
    _Float16* h1_f = (_Float16*)(ws + 310902784);
    _Float16* h1_b = (_Float16*)(ws + 311164928);
    float*    c_f  = (float*)   (ws + 311427072);      // 512 KB each
    float*    c_b  = (float*)   (ws + 311951360);

    // 1) Convert x to f16
    {
        const int n = BB * TSEQ * DD;                  // 16,777,216
        k_f32_to_f16<<<n / 256, 256, 0, stream>>>(x, x16, n);
    }
    // 2) Transpose + convert the four [512,2048] weight matrices to [2048,512] f16
    {
        dim3 g(G4 / 16, DD / 16), b(16, 16);
        k_transpose_f16<<<g, b, 0, stream>>>(W_f, wt_f);
        k_transpose_f16<<<g, b, 0, stream>>>(W_b, wt_b);
        k_transpose_f16<<<g, b, 0, stream>>>(U_f, ut_f);
        k_transpose_f16<<<g, b, 0, stream>>>(U_b, ut_b);
    }
    // 3) Zero initial h (h0_f+h0_b contiguous) and c (c_f+c_b contiguous)
    {
        const int n = 2 * BB * HH;                     // 262,144
        k_zero_state<<<n / 256, 256, 0, stream>>>(h0_f, c_f, n);
    }
    // 4) Input projection GEMM for both directions
    {
        dim3 g(G4 / 512, (BB * TSEQ) / 32, 2);         // (4, 1024, 2)
        k_inproj<<<g, 256, 0, stream>>>(x16, wt_f, wt_b, b_f, b_b, xw_f, xw_b);
    }
    // 5) 128 recurrent steps, double-buffered h
    {
        dim3 g(HH / 128, BB / 32, 2);                  // (4, 8, 2)
        for (int s = 0; s < TSEQ; ++s) {
            const bool even = (s & 1) == 0;
            const _Float16* hin_f  = even ? h0_f : h1_f;
            const _Float16* hin_b  = even ? h0_b : h1_b;
            _Float16*       hout_f = even ? h1_f : h0_f;
            _Float16*       hout_b = even ? h1_b : h0_b;
            k_step<<<g, 256, 0, stream>>>(ut_f, ut_b, xw_f, xw_b,
                                          hin_f, hin_b, hout_f, hout_b,
                                          c_f, c_b, out, s);
        }
    }
}
